// PredNoiseAnd3DLUT_62302795595973
// MI455X (gfx1250) — compile-verified, matrix operand order checked by reading, not run on hardware
//
#include <hip/hip_runtime.h>
#include <math.h>

// ---------------------------------------------------------------------------
// MI455X (gfx1250) implementation.
//   * lb_w1 conv  -> implicit-GEMM with v_wmma_f32_16x16x32_bf16 (bf16 in /
//     f32 acc). K ordered tap-major so each 32-k step sits inside one 3x3 tap
//     (no div/mod in the hot loop). 8 waves x 2 N-tiles, LDS double-buffered
//     A tile => 1 barrier + 2 WMMAs per k-step. Fused bias+ReLU+mean-pool.
//   * lut GEMM    -> HBM-bandwidth streaming (552 MB of lb_w2), wave/row,
//     output transposed for the trilinear gather stage.
//   * everything else is latency-bound VALU work on tiny tensors.
// ---------------------------------------------------------------------------

typedef unsigned short u16;
typedef __bf16 v16bf __attribute__((ext_vector_type(16)));
typedef float  v8f   __attribute__((ext_vector_type(8)));

#define DLUT 33
#define D3   35937
#define NOUT (15 * D3)          /* 539055 */
#define CINCH 320

// workspace layout (bytes)
#define FEAT_OFF   0u           /* 1,310,720 bf16 = 2,621,440 B */
#define WPACK_OFF  2621440u     /*   737,280 bf16 = 1,474,560 B */
#define POOLED_OFF 4096000u     /* 1024 f32 */
#define LUTS_OFF   4100096u     /* 4*35937*16 f32 = 9,199,872 B */
#define RBUF0_OFF  13299968u    /* 1,310,720 f32 */
#define RBUF1_OFF  18542848u    /* 1,310,720 f32 */

__device__ __forceinline__ u16 f2bf(float f) {
  unsigned u = __float_as_uint(f);
  unsigned r = u + 0x7FFFu + ((u >> 16) & 1u);   // round-to-nearest-even
  return (u16)(r >> 16);
}

// ---------------------------------------------------------------- packing ---
__global__ void pack_feat_bf16(const float* __restrict__ src,
                               u16* __restrict__ dst, int n) {
  int i = blockIdx.x * blockDim.x + threadIdx.x;
  if (i < n) dst[i] = f2bf(src[i]);
}

// Pack lb_w1 (256,320,3,3) into exact B-fragment order with tap-major K:
// [ntile][kstep][lane][e], k = kstep*32 + 16*(e>=8) + 8*(lane>=16) + (e&7),
// k -> (tap = k/320, ci = k%320), n = ntile*16 + (lane&15).
__global__ void pack_w_bf16(const float* __restrict__ w, u16* __restrict__ dst) {
  int t = blockIdx.x * blockDim.x + threadIdx.x;
  if (t >= 16 * 90 * 512) return;
  int e     = t & 15;
  int lane  = (t >> 4) & 31;
  int kstep = (t >> 9) % 90;
  int ntile = t / (90 * 512);
  int n     = (ntile << 4) + (lane & 15);
  int half  = lane >> 4;
  int k     = (kstep << 5) + ((e >> 3) << 4) + (half << 3) + (e & 7);
  int tap   = k / CINCH;           // tap-major K order
  int ci    = k - tap * CINCH;
  dst[t] = f2bf(w[(n * CINCH + ci) * 9 + tap]);
}

// ------------------------------------------------- WMMA conv + mean pool ---
// grid = 256 M-tiles (16 pixels each), block = 256 threads = 8 waves,
// wave w owns output channels [32w, 32w+32) as two 16x16 WMMA tiles.
// A tile double-buffered in LDS: one barrier, two v_wmma per k-step.
__global__ __launch_bounds__(256) void conv_wmma_pool(
    const u16* __restrict__ featbf, const u16* __restrict__ wpack,
    const float* __restrict__ bias, float* __restrict__ pooled) {
  __shared__ __align__(16) u16 sA[2][512];
  const int tid  = threadIdx.x;
  const int wave = tid >> 5;               // 0..7
  const int lane = tid & 31;
  const int pbase = blockIdx.x << 4;       // first pixel of this M-tile
  const int b = pbase >> 10;               // 1024 pixels per batch image

  // A-fill: thread fills elements tid (half=0) and tid+256 (half=1); both map
  // to the same pixel m and same tap, ci offsets kofs and kofs+8.
  const int e = tid & 15;
  const int m = (tid >> 4) & 15;
  const int kofs = ((e >> 3) << 4) + (e & 7);      // half=0 k-offset in [0,32)
  const int p  = pbase + m;
  const int py = (p >> 5) & 31, px = p & 31;
  const int fbase = (b * CINCH) << 10;

  v8f acc0 = {}, acc1 = {};
  const int nt0 = wave << 1;
  const u16* wp0 = wpack + ((nt0 * 90) << 9) + (lane << 4);
  const u16* wp1 = wp0 + (90 << 9);

#pragma unroll
  for (int tap = 0; tap < 9; ++tap) {
    const int dy = tap / 3, dx = tap - dy * 3;     // constants after unroll
    const int yy = py + dy - 1, xx = px + dx - 1;
    const bool valid = ((unsigned)yy < 32u) && ((unsigned)xx < 32u);
    const int gofs = fbase + (yy << 5) + xx;
    for (int sub = 0; sub < 10; ++sub) {           // 10 k-steps per tap
      const int buf = sub & 1;                     // tap*10 is even -> ok
      const int ci = (sub << 5) + kofs;
      u16 v0 = 0, v1 = 0;
      if (valid) {
        v0 = featbf[gofs + (ci << 10)];
        v1 = featbf[gofs + ((ci + 8) << 10)];
      }
      sA[buf][tid]       = v0;
      sA[buf][tid + 256] = v1;
      __syncthreads();

      union { uint4 q[2]; v16bf v; } fa, fb0, fb1;
      const uint4* pa = (const uint4*)(&sA[buf][lane << 4]);
      fa.q[0] = pa[0]; fa.q[1] = pa[1];            // ds_load_b128 x2
      const uint4* pb0 = (const uint4*)wp0;
      fb0.q[0] = pb0[0]; fb0.q[1] = pb0[1];        // global_load_b128 x2
      const uint4* pb1 = (const uint4*)wp1;
      fb1.q[0] = pb1[0]; fb1.q[1] = pb1[1];

      acc0 = __builtin_amdgcn_wmma_f32_16x16x32_bf16(
          false, fa.v, false, fb0.v, (short)0, acc0, false, false);
      acc1 = __builtin_amdgcn_wmma_f32_16x16x32_bf16(
          false, fa.v, false, fb1.v, (short)0, acc1, false, false);

      wp0 += 512; wp1 += 512;
      __builtin_prefetch(wp0, 0, 0);               // global_prefetch_b8
      __builtin_prefetch(wp1, 0, 0);
    }
  }

  // epilogue: bias + ReLU + partial mean over this tile's 16 pixels
  int nb = (wave << 5) + (lane & 15);
  float bv0 = bias[nb], bv1 = bias[nb + 16];
  float s0 = 0.f, s1 = 0.f;
#pragma unroll
  for (int i = 0; i < 8; ++i) {
    float h0 = acc0[i] + bv0;
    float h1 = acc1[i] + bv1;
    s0 += h0 > 0.f ? h0 : 0.f;
    s1 += h1 > 0.f ? h1 : 0.f;
  }
  s0 += __shfl_xor(s0, 16, 32);                    // combine pixel halves
  s1 += __shfl_xor(s1, 16, 32);
  if (lane < 16) {
    atomicAdd(&pooled[(b << 8) + nb],      s0 * (1.0f / 1024.0f));
    atomicAdd(&pooled[(b << 8) + nb + 16], s1 * (1.0f / 1024.0f));
  }
}

// ------------------------------------------------ lut GEMM (BW streaming) ---
// one wave per row of lb_w2 (1 KB contiguous); writes transposed lut table
// lutst[b][voxel][kc] (16-float padded rows) for locality in the sampler.
__global__ __launch_bounds__(256) void lut_gemm(
    const float* __restrict__ w2, const float* __restrict__ b2,
    const float* __restrict__ pooled, float* __restrict__ lutst) {
  __shared__ float sp[1024];
  const int tid = threadIdx.x;
  for (int i = tid; i < 1024; i += 256) sp[i] = pooled[i];
  __syncthreads();

  long row = (long)blockIdx.x * 8 + (tid >> 5);
  if (row >= NOUT) return;
  const int lane = tid & 31;
  const float* wr = w2 + row * 256 + (lane << 3);
  float4 a0 = ((const float4*)wr)[0];
  float4 a1 = ((const float4*)wr)[1];
  float s[4];
#pragma unroll
  for (int bb = 0; bb < 4; ++bb) {
    const float* pb = sp + (bb << 8) + (lane << 3);
    s[bb] = a0.x * pb[0] + a0.y * pb[1] + a0.z * pb[2] + a0.w * pb[3] +
            a1.x * pb[4] + a1.y * pb[5] + a1.z * pb[6] + a1.w * pb[7];
  }
#pragma unroll
  for (int off = 16; off; off >>= 1)
#pragma unroll
    for (int bb = 0; bb < 4; ++bb) s[bb] += __shfl_xor(s[bb], off, 32);

  if (lane == 0) {
    int kc  = (int)(row / D3);
    int idx = (int)(row - (long)kc * D3);
    float bias = b2[row];
#pragma unroll
    for (int bb = 0; bb < 4; ++bb)
      lutst[(((long)bb * D3 + idx) << 4) + kc] = s[bb] + bias;
  }
}

// ------------------------------------------------------------ noise conv ---
__global__ void noise_conv(const float* __restrict__ feat,
                           const float* __restrict__ w,
                           const float* __restrict__ bias,
                           float* __restrict__ out) {
  int gid = blockIdx.x * blockDim.x + threadIdx.x;
  if (gid >= 4 * 4 * 1024) return;
  int b = gid >> 12, co = (gid >> 10) & 3;
  int y = (gid >> 5) & 31, x = gid & 31;
  float s = bias[co];
  for (int dy = 0; dy < 3; ++dy) {
    int yy = y + dy - 1; if ((unsigned)yy >= 32u) continue;
    for (int dx = 0; dx < 3; ++dx) {
      int xx = x + dx - 1; if ((unsigned)xx >= 32u) continue;
      const float* fp = feat + ((b * CINCH) << 10) + (yy << 5) + xx;
      const float* wp = w + (co * CINCH) * 9 + dy * 3 + dx;
      for (int ci = 0; ci < CINCH; ++ci) s += fp[ci << 10] * wp[ci * 9];
    }
  }
  out[gid] = s;
}

// --------------------------------------------------------------- r branch ---
__global__ void r0_conv(const float* __restrict__ feat,
                        const float* __restrict__ w,
                        const float* __restrict__ bias,
                        float* __restrict__ out) {
  int gid = blockIdx.x * blockDim.x + threadIdx.x;
  if (gid >= 4 * 5 * 1024) return;
  int b = gid / 5120, co = (gid >> 10) % 5, pix = gid & 1023;
  float s = bias[co];
  const float* fp = feat + ((b * CINCH) << 10) + pix;
  const float* wp = w + co * CINCH;
  for (int ci = 0; ci < CINCH; ++ci) s += fp[ci << 10] * wp[ci];
  out[gid] = fmaxf(s, 0.f);
}

__global__ void conv3_k5(const float* __restrict__ in, const float* __restrict__ w,
                         const float* __restrict__ bias, float* __restrict__ out,
                         int S) {
  int gid = blockIdx.x * blockDim.x + threadIdx.x;
  int ss = S * S;
  if (gid >= 20 * ss) return;
  int b = gid / (5 * ss);
  int rem = gid - b * 5 * ss;
  int co = rem / ss;
  int pix = rem - co * ss;
  int y = pix / S, x = pix - y * S;
  float s = bias[co];
  for (int ky = 0; ky < 3; ++ky) {
    int yy = y + ky - 1; if ((unsigned)yy >= (unsigned)S) continue;
    for (int kx = 0; kx < 3; ++kx) {
      int xx = x + kx - 1; if ((unsigned)xx >= (unsigned)S) continue;
#pragma unroll
      for (int ci = 0; ci < 5; ++ci)
        s += in[(b * 5 + ci) * ss + yy * S + xx] *
             w[(co * 5 + ci) * 9 + ky * 3 + kx];
    }
  }
  out[gid] = fmaxf(s, 0.f);
}

// ConvTranspose2d(k=4, s=2, p=1): out[oy] gathers x[(oy+ky-2)/2] when even,
// weight = w[ci][co][3-ky][3-kx]  (flip+transpose of the reference).
__global__ void tconv4_k5(const float* __restrict__ in, const float* __restrict__ w,
                          const float* __restrict__ bias, float* __restrict__ out,
                          int S) {
  int O = S * 2, os = O * O;
  int gid = blockIdx.x * blockDim.x + threadIdx.x;
  if (gid >= 20 * os) return;
  int b = gid / (5 * os);
  int rem = gid - b * 5 * os;
  int co = rem / os;
  int pix = rem - co * os;
  int oy = pix / O, ox = pix - oy * O;
  float s = bias[co];
  for (int ky = 0; ky < 4; ++ky) {
    int ty = oy + ky - 2;
    if (ty < 0 || (ty & 1)) continue;
    int iy = ty >> 1; if (iy >= S) continue;
    for (int kx = 0; kx < 4; ++kx) {
      int tx = ox + kx - 2;
      if (tx < 0 || (tx & 1)) continue;
      int ix = tx >> 1; if (ix >= S) continue;
#pragma unroll
      for (int ci = 0; ci < 5; ++ci)
        s += in[(b * 5 + ci) * S * S + iy * S + ix] *
             w[((ci * 5 + co) << 4) + (3 - ky) * 4 + (3 - kx)];
    }
  }
  out[gid] = s;
}

// ---------------------------- fused trilinear sample + softmax + combine ---
__global__ __launch_bounds__(256) void fuse_lut(
    const float* __restrict__ img, const float* __restrict__ rlog,
    const float* __restrict__ lutst, float* __restrict__ out) {
  int gid = blockIdx.x * blockDim.x + threadIdx.x;
  if (gid >= 4 * 65536) return;
  int b = gid >> 16, pix = gid & 65535;

  int i0[3], i1[3]; float tf[3];
#pragma unroll
  for (int c = 0; c < 3; ++c) {
    float v = img[((b * 3 + c) << 16) + pix] * 0.5f + 0.5f;
    float f = fminf(fmaxf(v * 32.0f, 0.0f), 32.0f);
    int ii = (int)floorf(f);
    if (ii > 32) ii = 32;
    i0[c] = ii;
    tf[c] = f - (float)ii;
    i1[c] = (ii + 1 > 32) ? 32 : ii + 1;
  }

  float lg[5]; float mx = -1e30f;
#pragma unroll
  for (int k = 0; k < 5; ++k) {
    lg[k] = rlog[((b * 5 + k) << 16) + pix];
    mx = fmaxf(mx, lg[k]);
  }
  float se = 0.f;
#pragma unroll
  for (int k = 0; k < 5; ++k) { lg[k] = __expf(lg[k] - mx); se += lg[k]; }
  float inv = 1.0f / se;
#pragma unroll
  for (int k = 0; k < 5; ++k) lg[k] *= inv;

  float acc0 = 0.f, acc1 = 0.f, acc2 = 0.f;
#pragma unroll
  for (int dz = 0; dz < 2; ++dz) {
    int z = dz ? i1[2] : i0[2]; float wz = dz ? tf[2] : 1.f - tf[2];
#pragma unroll
    for (int dy = 0; dy < 2; ++dy) {
      int yc = dy ? i1[1] : i0[1]; float wy = dy ? tf[1] : 1.f - tf[1];
#pragma unroll
      for (int dx = 0; dx < 2; ++dx) {
        int xc = dx ? i1[0] : i0[0]; float wx = dx ? tf[0] : 1.f - tf[0];
        float w8 = wz * wy * wx;
        const float* T =
            lutst + (((long)b * D3 + ((z * 33 + yc) * 33 + xc)) << 4);
        float4 t0 = ((const float4*)T)[0];
        float4 t1 = ((const float4*)T)[1];
        float4 t2 = ((const float4*)T)[2];
        float4 t3 = ((const float4*)T)[3];
        float tv[16] = {t0.x, t0.y, t0.z, t0.w, t1.x, t1.y, t1.z, t1.w,
                        t2.x, t2.y, t2.z, t2.w, t3.x, t3.y, t3.z, t3.w};
#pragma unroll
        for (int k = 0; k < 5; ++k) {
          float wk = lg[k] * w8;
          acc0 += wk * tv[k * 3 + 0];
          acc1 += wk * tv[k * 3 + 1];
          acc2 += wk * tv[k * 3 + 2];
        }
      }
    }
  }
  float* o = out + 16384 + ((b * 3) << 16) + pix;
  o[0] = acc0;
  o[1 << 16] = acc1;
  o[2 << 16] = acc2;
}

// ---------------------------------------------------------------------------
extern "C" void kernel_launch(void* const* d_in, const int* in_sizes, int n_in,
                              void* d_out, int out_size, void* d_ws,
                              size_t ws_size, hipStream_t stream) {
  (void)in_sizes; (void)n_in; (void)out_size; (void)ws_size;
  const float* feat       = (const float*)d_in[0];
  const float* img        = (const float*)d_in[1];
  const float* conv_out_w = (const float*)d_in[2];
  const float* conv_out_b = (const float*)d_in[3];
  const float* lb_w1      = (const float*)d_in[4];
  const float* lb_b1      = (const float*)d_in[5];
  const float* lb_w2      = (const float*)d_in[6];
  const float* lb_b2      = (const float*)d_in[7];
  const float* rc0w = (const float*)d_in[8];
  const float* rc0b = (const float*)d_in[9];
  const float* rc1w = (const float*)d_in[10];
  const float* rc1b = (const float*)d_in[11];
  const float* rt0w = (const float*)d_in[12];
  const float* rt0b = (const float*)d_in[13];
  const float* rc2w = (const float*)d_in[14];
  const float* rc2b = (const float*)d_in[15];
  const float* rt1w = (const float*)d_in[16];
  const float* rt1b = (const float*)d_in[17];
  const float* rc3w = (const float*)d_in[18];
  const float* rc3b = (const float*)d_in[19];
  const float* rt2w = (const float*)d_in[20];
  const float* rt2b = (const float*)d_in[21];

  float* out = (float*)d_out;
  char* ws = (char*)d_ws;
  u16*   featbf = (u16*)(ws + FEAT_OFF);
  u16*   wpack  = (u16*)(ws + WPACK_OFF);
  float* pooled = (float*)(ws + POOLED_OFF);
  float* lutst  = (float*)(ws + LUTS_OFF);
  float* rbuf0  = (float*)(ws + RBUF0_OFF);
  float* rbuf1  = (float*)(ws + RBUF1_OFF);

  hipMemsetAsync(pooled, 0, 1024 * sizeof(float), stream);
  pack_feat_bf16<<<5120, 256, 0, stream>>>(feat, featbf, 4 * CINCH * 1024);
  pack_w_bf16<<<2880, 256, 0, stream>>>(lb_w1, wpack);
  conv_wmma_pool<<<256, 256, 0, stream>>>(featbf, wpack, lb_b1, pooled);
  lut_gemm<<<(NOUT + 7) / 8, 256, 0, stream>>>(lb_w2, lb_b2, pooled, lutst);
  noise_conv<<<64, 256, 0, stream>>>(feat, conv_out_w, conv_out_b, out);

  r0_conv<<<80, 256, 0, stream>>>(feat, rc0w, rc0b, rbuf0);
  conv3_k5<<<80, 256, 0, stream>>>(rbuf0, rc1w, rc1b, rbuf1, 32);
  tconv4_k5<<<320, 256, 0, stream>>>(rbuf1, rt0w, rt0b, rbuf0, 32);
  conv3_k5<<<320, 256, 0, stream>>>(rbuf0, rc2w, rc2b, rbuf1, 64);
  tconv4_k5<<<1280, 256, 0, stream>>>(rbuf1, rt1w, rt1b, rbuf0, 64);
  conv3_k5<<<1280, 256, 0, stream>>>(rbuf0, rc3w, rc3b, rbuf1, 128);
  tconv4_k5<<<5120, 256, 0, stream>>>(rbuf1, rt2w, rt2b, rbuf0, 128);

  fuse_lut<<<1024, 256, 0, stream>>>(img, rbuf0, lutst, out);
}
